// Net_17214228922567
// MI455X (gfx1250) — matrix operand order repaired and checked by reading, not compiled, not run on hardware
//
#include <hip/hip_runtime.h>
#include <hip/hip_bf16.h>
#include <cmath>

// ---------------------------------------------------------------------------
// Model dims (compile-time constants from the reference)
// ---------------------------------------------------------------------------
#define BB       8
#define LL       1024
#define IN_DIM   64
#define D_MODEL  768
#define D_INNER  1536
#define D_STATE  16
#define D_CONV   4
#define DT_RANK  48
#define DT_PAD   64          // DT_RANK padded up to WMMA K granularity
#define N_LAYERS 2
#define MTOK     (BB * LL)   // 8192 tokens

// ---------------------------------------------------------------------------
// bf16 helpers (storage = unsigned short, compute via __bf16 ext vectors)
// ---------------------------------------------------------------------------
typedef __bf16 bf16_t;
typedef bf16_t v16bf __attribute__((ext_vector_type(16)));
typedef bf16_t v8bf  __attribute__((ext_vector_type(8)));
typedef float  v8f   __attribute__((ext_vector_type(8)));

__device__ __forceinline__ unsigned short f2bf(float f) {
    union { float f; unsigned u; } v; v.f = f;
    unsigned r = v.u + 0x7FFFu + ((v.u >> 16) & 1u);   // round-to-nearest-even
    return (unsigned short)(r >> 16);
}

__device__ __forceinline__ float silu_f(float v) {
    return v / (1.0f + __expf(-v));
}

// ---------------------------------------------------------------------------
// Generic fp32 -> bf16 conversion
// ---------------------------------------------------------------------------
__global__ void cvt_f32_bf16(const float* __restrict__ in,
                             unsigned short* __restrict__ out, int n) {
    int i = blockIdx.x * blockDim.x + threadIdx.x;
    if (i < n) out[i] = f2bf(in[i]);
}

// fp32 (rows x incols, leading dim in_ld) -> bf16 (rows x outcols, zero-padded)
__global__ void cvt_pad_bf16(const float* __restrict__ in,
                             unsigned short* __restrict__ out,
                             int rows, int in_ld, int incols, int outcols) {
    long i = (long)blockIdx.x * blockDim.x + threadIdx.x;
    long n = (long)rows * outcols;
    if (i >= n) return;
    int  c = (int)(i % outcols);
    long r = i / outcols;
    out[i] = (c < incols) ? f2bf(in[r * in_ld + c]) : (unsigned short)0;
}

// ---------------------------------------------------------------------------
// WMMA bf16 GEMM:  C[M,N] (fp32) = A[M,K] (bf16, row-major, ld=lda)
//                                x W[N,K]^T (bf16, row-major, ld=ldw)
// Each wave computes a (16*MT) x (16*NT) tile of C via
// v_wmma_f32_16x16x32_bf16; register blocking MT x NT reuses each B fragment
// MT times and each A fragment NT times (0.75 KB moved per WMMA at 2x4).
// EPI: 0 = none, 1 = +bias, 2 = softplus(acc + bias), 3 = +resid (residual add)
// ---------------------------------------------------------------------------
template <int MT, int NT, int EPI>
__launch_bounds__(256)
__global__ void gemm_bf16_wmma(const unsigned short* __restrict__ Abuf,
                               const unsigned short* __restrict__ Wbuf,
                               float* __restrict__ C,
                               const float* __restrict__ bias,
                               const float* __restrict__ resid,
                               int M, int N, int K, int lda, int ldw, int ldc) {
    const int lane = threadIdx.x & 31;
    const int gw   = blockIdx.x * (blockDim.x >> 5) + (threadIdx.x >> 5);
    const int ntiles = N / (16 * NT);
    const int mt = gw / ntiles;
    const int nt = gw % ntiles;
    if (mt * 16 * MT >= M) return;
    const int m0 = mt * 16 * MT;
    const int n0 = nt * 16 * NT;

    const int half = lane >> 4;   // 0: K 0..7 / 16..23   1: K 8..15 / 24..31
    const int l16  = lane & 15;

    const bf16_t* A = (const bf16_t*)Abuf;
    const bf16_t* W = (const bf16_t*)Wbuf;

    // A fragment bases: row m0 + mi*16 + l16, two 8-wide chunks per K step
    const bf16_t* arow[MT];
#pragma unroll
    for (int mi = 0; mi < MT; ++mi)
        arow[mi] = A + (size_t)(m0 + mi * 16 + l16) * lda + half * 8;

    v8f acc[MT][NT];
#pragma unroll
    for (int mi = 0; mi < MT; ++mi)
#pragma unroll
        for (int j = 0; j < NT; ++j)
#pragma unroll
            for (int e = 0; e < 8; ++e) acc[mi][j][e] = 0.0f;

    for (int k0 = 0; k0 < K; k0 += 32) {
        union { v16bf v; v8bf h[2]; } a[MT];
#pragma unroll
        for (int mi = 0; mi < MT; ++mi) {
            a[mi].h[0] = *(const v8bf*)(arow[mi] + k0);       // K k0 + half*8 ..
            a[mi].h[1] = *(const v8bf*)(arow[mi] + k0 + 16);  // K k0+16+half*8 ..
        }
        v16bf b[NT];
#pragma unroll
        for (int j = 0; j < NT; ++j) {
            // B column (n0+16j+l16) == W row; lane half selects K 0..7,16..23 / 8..15,24..31
            const bf16_t* wrow =
                W + (size_t)(n0 + j * 16 + l16) * ldw + half * 8 + k0;
            union { v16bf v; v8bf h[2]; } bt;
            bt.h[0] = *(const v8bf*)wrow;
            bt.h[1] = *(const v8bf*)(wrow + 16);
            b[j] = bt.v;
        }
#pragma unroll
        for (int mi = 0; mi < MT; ++mi)
#pragma unroll
            for (int j = 0; j < NT; ++j)
                acc[mi][j] = __builtin_amdgcn_wmma_f32_16x16x32_bf16(
                    false, a[mi].v, false, b[j], (short)0, acc[mi][j],
                    false, false);
    }

    // Epilogue: VGPR r holds (m = m0 + mi*16 + half*8 + r, n = n0 + 16j + l16)
#pragma unroll
    for (int j = 0; j < NT; ++j) {
        const int col = n0 + j * 16 + l16;
        float bv = 0.0f;
        if (EPI == 1 || EPI == 2) bv = bias[col];
#pragma unroll
        for (int mi = 0; mi < MT; ++mi) {
#pragma unroll
            for (int r = 0; r < 8; ++r) {
                const int m = m0 + mi * 16 + half * 8 + r;
                float v = acc[mi][j][r];
                if (EPI == 1) {
                    v += bv;
                } else if (EPI == 2) {            // softplus(v + bias)
                    v += bv;
                    v = (v > 20.0f) ? v : log1pf(__expf(v));
                } else if (EPI == 3) {
                    v += resid[(size_t)m * ldc + col];
                }
                C[(size_t)m * ldc + col] = v;
            }
        }
    }
}

// ---------------------------------------------------------------------------
// RMSNorm over D_MODEL, fused bf16 output. One block per token.
// ---------------------------------------------------------------------------
__global__ void rmsnorm_kernel(const float* __restrict__ h,
                               const float* __restrict__ w,
                               unsigned short* __restrict__ out, int D) {
    __shared__ float red[256];
    const int t = blockIdx.x;
    const float* row = h + (size_t)t * D;
    float s = 0.0f;
    for (int c = threadIdx.x; c < D; c += blockDim.x) {
        float v = row[c];
        s += v * v;
    }
    red[threadIdx.x] = s;
    __syncthreads();
    for (int k = 128; k > 0; k >>= 1) {
        if (threadIdx.x < k) red[threadIdx.x] += red[threadIdx.x + k];
        __syncthreads();
    }
    const float scale = rsqrtf(red[0] / (float)D + 1e-5f);
    for (int c = threadIdx.x; c < D; c += blockDim.x)
        out[(size_t)t * D + c] = f2bf(row[c] * scale * w[c]);
}

// ---------------------------------------------------------------------------
// Causal depthwise conv (D_CONV=4) + bias + SiLU over xb (= xz cols 0..D_INNER)
// Writes fp32 xc (scan input u) and bf16 xb16 (x_proj GEMM input).
// ---------------------------------------------------------------------------
__global__ void conv_silu_kernel(const float* __restrict__ xz,
                                 const float* __restrict__ cw,
                                 const float* __restrict__ cb,
                                 float* __restrict__ xc,
                                 unsigned short* __restrict__ xb16) {
    long i = (long)blockIdx.x * blockDim.x + threadIdx.x;
    const long n = (long)MTOK * D_INNER;
    if (i >= n) return;
    const int d  = (int)(i % D_INNER);
    const long bl = i / D_INNER;
    const int l  = (int)(bl % LL);
    const int b  = (int)(bl / LL);
    const float* base = xz + ((size_t)b * LL) * (2 * D_INNER) + d;
    float s = 0.0f;
#pragma unroll
    for (int t = 0; t < D_CONV; ++t) {
        const int ls = l - (D_CONV - 1) + t;
        if (ls >= 0) s += cw[d * D_CONV + t] * base[(size_t)ls * (2 * D_INNER)];
    }
    s += cb[d];
    const float sv = silu_f(s);
    xc[i]   = sv;
    xb16[i] = f2bf(sv);
}

// ---------------------------------------------------------------------------
// Selective scan: one thread per (b, d) channel, 16-state register recurrence.
// y written strided into the xb half of xz (ld = 2*D_INNER).
// ---------------------------------------------------------------------------
__global__ void scan_kernel(const float* __restrict__ delta,
                            const float* __restrict__ u,
                            const float* __restrict__ dbc,
                            const float* __restrict__ A_log,
                            const float* __restrict__ Dp,
                            float* __restrict__ y, int ldy) {
    const int i = blockIdx.x * blockDim.x + threadIdx.x;
    if (i >= BB * D_INNER) return;
    const int d = i % D_INNER;
    const int b = i / D_INNER;

    float Arow[D_STATE];
#pragma unroll
    for (int s = 0; s < D_STATE; ++s) Arow[s] = -__expf(A_log[d * D_STATE + s]);
    const float Dv = Dp[d];

    float hst[D_STATE];
#pragma unroll
    for (int s = 0; s < D_STATE; ++s) hst[s] = 0.0f;

    for (int l = 0; l < LL; ++l) {
        const size_t idx = (size_t)b * LL + l;
        const float dt = delta[idx * D_INNER + d];
        const float ut = u[idx * D_INNER + d];
        const float du = dt * ut;
        const float* bm = dbc + idx * (DT_RANK + 2 * D_STATE) + DT_RANK;
        const float* cm = bm + D_STATE;
        float yv = 0.0f;
#pragma unroll
        for (int s = 0; s < D_STATE; ++s) {
            hst[s] = __expf(dt * Arow[s]) * hst[s] + du * bm[s];
            yv += hst[s] * cm[s];
        }
        y[idx * ldy + d] = yv + ut * Dv;
    }
}

// ---------------------------------------------------------------------------
// Gating: y16 = bf16( y * silu(z) ); y/z strided inside xz.
// ---------------------------------------------------------------------------
__global__ void gate_kernel(const float* __restrict__ xz,
                            unsigned short* __restrict__ y16) {
    long i = (long)blockIdx.x * blockDim.x + threadIdx.x;
    const long n = (long)MTOK * D_INNER;
    if (i >= n) return;
    const int  d   = (int)(i % D_INNER);
    const long idx = i / D_INNER;
    const float yv = xz[idx * (2 * D_INNER) + d];
    const float zv = xz[idx * (2 * D_INNER) + D_INNER + d];
    y16[i] = f2bf(yv * silu_f(zv));
}

// ---------------------------------------------------------------------------
// Final projection to OUT_DIM = 1: one thread per token.
// ---------------------------------------------------------------------------
__global__ void lin2_kernel(const float* __restrict__ h,
                            const float* __restrict__ w,
                            const float* __restrict__ b,
                            float* __restrict__ out) {
    const int t = blockIdx.x * blockDim.x + threadIdx.x;
    if (t >= MTOK) return;
    const float* row = h + (size_t)t * D_MODEL;
    float s = 0.0f;
    for (int k = 0; k < D_MODEL; ++k) s += row[k] * w[k];
    out[t] = s + b[0];
}

// ---------------------------------------------------------------------------
// Host-side orchestration
// ---------------------------------------------------------------------------
extern "C" void kernel_launch(void* const* d_in, const int* in_sizes, int n_in,
                              void* d_out, int out_size, void* d_ws, size_t ws_size,
                              hipStream_t stream) {
    (void)n_in; (void)out_size; (void)ws_size;

    // ---- resolve input pointers (insertion-order vs jax alphabetical) ----
    const float *x, *lin1_w, *lin1_b, *lin2_w, *lin2_b;
    struct LayerP {
        const float *norm_w, *in_proj_w, *conv_w, *conv_b, *x_proj_w,
                    *dt_proj_w, *dt_proj_b, *A_log, *Dp, *out_proj_w;
    } L[N_LAYERS];

    const bool insertion = (in_sizes[0] == MTOK * IN_DIM);
    if (insertion) {
        x      = (const float*)d_in[0];
        lin1_w = (const float*)d_in[1];
        lin1_b = (const float*)d_in[2];
        lin2_w = (const float*)d_in[3];
        lin2_b = (const float*)d_in[4];
        for (int i = 0; i < N_LAYERS; ++i) {
            const int base = 5 + i * 10;
            L[i].norm_w     = (const float*)d_in[base + 0];
            L[i].in_proj_w  = (const float*)d_in[base + 1];
            L[i].conv_w     = (const float*)d_in[base + 2];
            L[i].conv_b     = (const float*)d_in[base + 3];
            L[i].x_proj_w   = (const float*)d_in[base + 4];
            L[i].dt_proj_w  = (const float*)d_in[base + 5];
            L[i].dt_proj_b  = (const float*)d_in[base + 6];
            L[i].A_log      = (const float*)d_in[base + 7];
            L[i].Dp         = (const float*)d_in[base + 8];
            L[i].out_proj_w = (const float*)d_in[base + 9];
        }
    } else {
        // jax pytree flatten: dict keys alphabetical; 'params' < 'x'
        for (int i = 0; i < N_LAYERS; ++i) {
            const int base = i * 10;
            L[i].A_log      = (const float*)d_in[base + 0];
            L[i].Dp         = (const float*)d_in[base + 1];
            L[i].conv_b     = (const float*)d_in[base + 2];
            L[i].conv_w     = (const float*)d_in[base + 3];
            L[i].dt_proj_b  = (const float*)d_in[base + 4];
            L[i].dt_proj_w  = (const float*)d_in[base + 5];
            L[i].in_proj_w  = (const float*)d_in[base + 6];
            L[i].norm_w     = (const float*)d_in[base + 7];
            L[i].out_proj_w = (const float*)d_in[base + 8];
            L[i].x_proj_w   = (const float*)d_in[base + 9];
        }
        lin1_b = (const float*)d_in[20];
        lin1_w = (const float*)d_in[21];
        lin2_b = (const float*)d_in[22];
        lin2_w = (const float*)d_in[23];
        x      = (const float*)d_in[24];
    }

    // ---- workspace carve (~275 MB total) ----
    char* ws = (char*)d_ws;
    size_t off = 0;
    auto carve = [&](size_t bytes) -> void* {
        void* p = ws + off;
        off = (off + bytes + 255) & ~(size_t)255;
        return p;
    };
    float*          h      = (float*)carve((size_t)MTOK * D_MODEL * 4);
    unsigned short* hn16   = (unsigned short*)carve((size_t)MTOK * D_MODEL * 2);
    float*          xz     = (float*)carve((size_t)MTOK * 2 * D_INNER * 4);
    float*          xc     = (float*)carve((size_t)MTOK * D_INNER * 4);
    unsigned short* act16  = (unsigned short*)carve((size_t)MTOK * D_INNER * 2);
    float*          dbc    = (float*)carve((size_t)MTOK * (DT_RANK + 2 * D_STATE) * 4);
    unsigned short* dpad   = (unsigned short*)carve((size_t)MTOK * DT_PAD * 2);
    float*          delta  = (float*)carve((size_t)MTOK * D_INNER * 4);
    unsigned short* w16_in = (unsigned short*)carve((size_t)2 * D_INNER * D_MODEL * 2);
    unsigned short* w16_x  = (unsigned short*)carve((size_t)(DT_RANK + 2 * D_STATE) * D_INNER * 2);
    unsigned short* w16_dt = (unsigned short*)carve((size_t)D_INNER * DT_PAD * 2);
    unsigned short* w16_o  = (unsigned short*)carve((size_t)D_MODEL * D_INNER * 2);
    unsigned short* x16    = (unsigned short*)carve((size_t)MTOK * IN_DIM * 2);
    unsigned short* w16_l1 = (unsigned short*)carve((size_t)D_MODEL * IN_DIM * 2);

    auto cvt = [&](const float* in, unsigned short* out, int n) {
        cvt_f32_bf16<<<(n + 255) / 256, 256, 0, stream>>>(in, out, n);
    };
    auto gemm_blocks = [](int M, int N, int MT, int NT) {
        const int waves = (M / (16 * MT)) * (N / (16 * NT));
        return (waves + 7) / 8;
    };

    const long nelem = (long)MTOK * D_INNER;
    const int ew_blocks = (int)((nelem + 255) / 256);

    // ================= lin1: h = x @ lin1_w.T + lin1_b =================
    cvt(x, x16, MTOK * IN_DIM);
    cvt(lin1_w, w16_l1, D_MODEL * IN_DIM);
    gemm_bf16_wmma<2, 4, 1><<<gemm_blocks(MTOK, D_MODEL, 2, 4), 256, 0, stream>>>(
        x16, w16_l1, h, lin1_b, nullptr,
        MTOK, D_MODEL, IN_DIM, IN_DIM, IN_DIM, D_MODEL);

    // ================= mamba layers =================
    for (int li = 0; li < N_LAYERS; ++li) {
        const LayerP& p = L[li];

        // rmsnorm -> bf16
        rmsnorm_kernel<<<MTOK, 256, 0, stream>>>(h, p.norm_w, hn16, D_MODEL);

        // in_proj: xz = hn @ in_proj_w.T   (8192 x 3072, K=768)
        cvt(p.in_proj_w, w16_in, 2 * D_INNER * D_MODEL);
        gemm_bf16_wmma<2, 4, 0><<<gemm_blocks(MTOK, 2 * D_INNER, 2, 4), 256, 0, stream>>>(
            hn16, w16_in, xz, nullptr, nullptr,
            MTOK, 2 * D_INNER, D_MODEL, D_MODEL, D_MODEL, 2 * D_INNER);

        // depthwise causal conv + bias + silu -> xc (fp32), act16 (bf16)
        conv_silu_kernel<<<ew_blocks, 256, 0, stream>>>(
            xz, p.conv_w, p.conv_b, xc, act16);

        // x_proj: dbc = xb @ x_proj_w.T    (8192 x 80, K=1536), NT=5
        cvt(p.x_proj_w, w16_x, (DT_RANK + 2 * D_STATE) * D_INNER);
        gemm_bf16_wmma<2, 5, 0><<<gemm_blocks(MTOK, DT_RANK + 2 * D_STATE, 2, 5), 256, 0, stream>>>(
            act16, w16_x, dbc, nullptr, nullptr,
            MTOK, DT_RANK + 2 * D_STATE, D_INNER, D_INNER, D_INNER,
            DT_RANK + 2 * D_STATE);

        // pad delta_raw (48 cols of dbc) and dt_proj_w (K 48->64) to bf16
        {
            const long n1 = (long)MTOK * DT_PAD;
            cvt_pad_bf16<<<(int)((n1 + 255) / 256), 256, 0, stream>>>(
                dbc, dpad, MTOK, DT_RANK + 2 * D_STATE, DT_RANK, DT_PAD);
            const long n2 = (long)D_INNER * DT_PAD;
            cvt_pad_bf16<<<(int)((n2 + 255) / 256), 256, 0, stream>>>(
                p.dt_proj_w, w16_dt, D_INNER, DT_RANK, DT_RANK, DT_PAD);
        }

        // dt_proj + softplus: delta = softplus(dpad @ dt_w.T + dt_b)
        gemm_bf16_wmma<2, 4, 2><<<gemm_blocks(MTOK, D_INNER, 2, 4), 256, 0, stream>>>(
            dpad, w16_dt, delta, p.dt_proj_b, nullptr,
            MTOK, D_INNER, DT_PAD, DT_PAD, DT_PAD, D_INNER);

        // selective scan -> y (strided into xb half of xz)
        scan_kernel<<<(BB * D_INNER + 255) / 256, 256, 0, stream>>>(
            delta, xc, dbc, p.A_log, p.Dp, xz, 2 * D_INNER);

        // gate: y16 = bf16( y * silu(z) )
        gate_kernel<<<ew_blocks, 256, 0, stream>>>(xz, act16);

        // out_proj + residual: h = h + y16 @ out_proj_w.T
        cvt(p.out_proj_w, w16_o, D_MODEL * D_INNER);
        gemm_bf16_wmma<2, 4, 3><<<gemm_blocks(MTOK, D_MODEL, 2, 4), 256, 0, stream>>>(
            act16, w16_o, h, nullptr, h,
            MTOK, D_MODEL, D_INNER, D_INNER, D_INNER, D_MODEL);
    }

    // ================= lin2: out = h @ lin2_w.T + lin2_b =================
    lin2_kernel<<<(MTOK + 255) / 256, 256, 0, stream>>>(
        h, lin2_w, lin2_b, (float*)d_out);
}
